// SS2D_54451595378864
// MI455X (gfx1250) — matrix operand order repaired and checked by reading, not compiled, not run on hardware
//
#include <hip/hip_runtime.h>
#include <math.h>

// ---------------------------------------------------------------------------
// SS2D selective scan for MI455X (gfx1250, wave32).
//   B=4, C=96, H=W=64 (L=4096), N=16, R=6, 4 scan paths.
//
// - Projections run on the WMMA pipe (v_wmma_f32_16x16x32_f16, f32 accum).
//   Weight (B-operand) tiles are packed into VGPRs ONCE per persistent wave;
//   each wave then streams 4 l-tiles doing only A-loads + WMMAs.
// - The sequential recurrence keeps its 16 states in VGPRs (lane = channel);
//   the shared B(t)/C(t) vectors are staged into LDS with
//   global_load_async_to_lds_b128 (ASYNCcnt) and the next chunk is hinted
//   with global_prefetch_b8.
// ---------------------------------------------------------------------------

#define BN   4          // batch
#define CD   96         // d_model
#define HH   64
#define WW   64
#define NS   16         // d_state
#define RR   6          // dt_rank
#define LL   (HH * WW)  // 4096
#define NPATH 4
#define KX   (RR + 2 * NS)   // 38 columns of x_dbl
#define LTILES (LL / 16)     // 256 l-tiles per (path,batch)
#define NTILES (NPATH * BN * LTILES)  // 4096 l-tiles total
#define NWAVES 1024                   // persistent waves for the GEMMs
#define TPW    (NTILES / NWAVES)      // 4 l-tiles per wave

typedef _Float16 v16h __attribute__((ext_vector_type(16)));
typedef float    v8f  __attribute__((ext_vector_type(8)));

// ---------------------------------------------------------------------------
// Async global->LDS staging (gfx1250): per-lane 16B copy tracked by ASYNCcnt.
// ---------------------------------------------------------------------------
__device__ __forceinline__ void async_copy_b128(const float* gaddr,
                                                float* lds_generic) {
  unsigned lds_addr =
      (unsigned)(unsigned long long)(__attribute__((address_space(3))) float*)
          lds_generic;
  asm volatile("global_load_async_to_lds_b128 %0, %1, off"
               :
               : "v"(lds_addr), "v"(gaddr)
               : "memory");
}
__device__ __forceinline__ void wait_asynccnt0() {
  asm volatile("s_wait_asynccnt 0x0" ::: "memory");
}

// ---------------------------------------------------------------------------
// Kernel 1: gather x into per-path scan order  U[(p,b,l),d]  (d contiguous).
// ---------------------------------------------------------------------------
__global__ __launch_bounds__(256) void ss2d_permute(const float* __restrict__ x,
                                                    float* __restrict__ U,
                                                    int total) {
  int idx = blockIdx.x * blockDim.x + threadIdx.x;
  if (idx >= total) return;
  int d = idx % CD;
  int t = idx / CD;
  int l = t % LL;
  t /= LL;
  int b = t % BN;
  int p = t / BN;
  int h, w;
  if (p == 0)      { h = l / WW; w = l % WW; }                 // raster
  else if (p == 1) { h = l / WW; w = (WW - 1) - (l % WW); }    // W flipped
  else if (p == 2) { w = l / HH; h = l % HH; }                 // column-major
  else             { w = l / HH; h = (HH - 1) - (l % HH); }    // H flipped
  U[idx] = x[((size_t)(b * CD + d) * HH + h) * WW + w];
}

// ---------------------------------------------------------------------------
// WMMA operand packing (wave32; A/B use contiguous k, not the ISA 7.12.2 table order):
//  A 16x32 f16 : m = lane&15, hi = lane>>4; VGPR v holds K pair
//                k0 = hi*16+2v
//  B 32x16 f16 : n = lane&15, hi = lane>>4; VGPR v holds K pair k0 = hi*16+2v
//  C/D 16x16 f32: n = lane&15; VGPR v -> row m = v + (lane>>4)*8
// ---------------------------------------------------------------------------

// Kernel 2: x_dbl = U @ Wx[p]  (per (p,b): L x 96 @ 96 x 38, N padded to 48).
// One persistent wave per 4 consecutive l-tiles (same (p,b)); the nine
// 32x16 B tiles (3 K-steps x 3 N-tiles) live in VGPRs across all tiles.
__global__ __launch_bounds__(256) void ss2d_xproj(const float* __restrict__ U,
                                                  const float* __restrict__ Wx,
                                                  float* __restrict__ dtb,
                                                  float* __restrict__ Bcb,
                                                  float* __restrict__ Ccb) {
  int wave = threadIdx.x >> 5;
  int lane = threadIdx.x & 31;
  int wgid = blockIdx.x * 8 + wave;        // 0 .. NWAVES-1
  int tile0 = wgid * TPW;                  // 4 tiles, all in the same (p,b)
  int pb = tile0 / LTILES;
  int p  = pb / BN;

  int m  = lane & 15;
  int hi = lane >> 4;
  int n  = lane & 15;

  const float* WxP = Wx + (size_t)p * CD * KX;

  // ---- pack the 9 weight tiles once (f32 -> f16)
  v16h Breg[3][3];
#pragma unroll
  for (int kki = 0; kki < 3; ++kki) {
#pragma unroll
    for (int t = 0; t < 3; ++t) {
      int col = t * 16 + n;
      v16h bm;
#pragma unroll
      for (int v = 0; v < 8; ++v) {
        int kd = kki * 32 + hi * 16 + 2 * v;
        float b0 = (col < KX) ? WxP[(size_t)kd * KX + col] : 0.f;
        float b1 = (col < KX) ? WxP[(size_t)(kd + 1) * KX + col] : 0.f;
        bm[2 * v]     = (_Float16)b0;
        bm[2 * v + 1] = (_Float16)b1;
      }
      Breg[kki][t] = bm;
    }
  }

  for (int tt = 0; tt < TPW; ++tt) {
    int tile = tile0 + tt;
    int lt = tile % LTILES;
    const float* Urow = U + ((size_t)pb * LL + (size_t)lt * 16 + m) * CD;

    v8f acc0 = {}, acc1 = {}, acc2 = {};
#pragma unroll
    for (int kki = 0; kki < 3; ++kki) {
      v16h a;
#pragma unroll
      for (int v = 0; v < 8; ++v) {
        int k0 = kki * 32 + hi * 16 + 2 * v;
        float2 f = *(const float2*)(Urow + k0);
        a[2 * v]     = (_Float16)f.x;
        a[2 * v + 1] = (_Float16)f.y;
      }
      acc0 = __builtin_amdgcn_wmma_f32_16x16x32_f16(false, a, false,
                                                    Breg[kki][0], (short)0,
                                                    acc0, false, false);
      acc1 = __builtin_amdgcn_wmma_f32_16x16x32_f16(false, a, false,
                                                    Breg[kki][1], (short)0,
                                                    acc1, false, false);
      acc2 = __builtin_amdgcn_wmma_f32_16x16x32_f16(false, a, false,
                                                    Breg[kki][2], (short)0,
                                                    acc2, false, false);
    }

    // ---- scatter the 16x48 tile into dt (k<6), Bc (6..21), Cc (22..37)
#pragma unroll
    for (int v = 0; v < 8; ++v) {
      size_t row = (size_t)pb * LL + (size_t)lt * 16 + v + hi * 8;
      float vals[3] = {acc0[v], acc1[v], acc2[v]};
#pragma unroll
      for (int t = 0; t < 3; ++t) {
        int k = t * 16 + n;
        float val = vals[t];
        if (k < RR)            dtb[row * RR + k] = val;
        else if (k < RR + NS)  Bcb[row * NS + (k - RR)] = val;
        else if (k < KX)       Ccb[row * NS + (k - RR - NS)] = val;
      }
    }
  }
}

// ---------------------------------------------------------------------------
// Kernel 3: delta = softplus(dt @ Wdt[p] + bdt[p])   (K padded 6 -> 32).
// Persistent waves again: six 32x16 B tiles + 6 biases resident in VGPRs.
// ---------------------------------------------------------------------------
__global__ __launch_bounds__(256) void ss2d_dtproj(const float* __restrict__ dtb,
                                                   const float* __restrict__ Wdt,
                                                   const float* __restrict__ bdt,
                                                   float* __restrict__ delta) {
  int wave = threadIdx.x >> 5;
  int lane = threadIdx.x & 31;
  int wgid = blockIdx.x * 8 + wave;
  int tile0 = wgid * TPW;
  int pb = tile0 / LTILES;
  int p  = pb / BN;

  int m  = lane & 15;
  int hi = lane >> 4;
  int n  = lane & 15;

  const float* WdtP = Wdt + (size_t)p * RR * CD;
  const float* bdtP = bdt + (size_t)p * CD;

  v16h Breg[6];
  float bias[6];
#pragma unroll
  for (int t = 0; t < 6; ++t) {
    int col = t * 16 + n;                 // output channel d (< 96)
    v16h bm;
#pragma unroll
    for (int v = 0; v < 8; ++v) {
      int kd = hi * 16 + 2 * v;           // K row, valid only kd<6
      float b0 = (kd < RR) ? WdtP[(size_t)kd * CD + col] : 0.f;
      float b1 = (kd < RR) ? WdtP[(size_t)(kd + 1) * CD + col] : 0.f;
      bm[2 * v]     = (_Float16)b0;
      bm[2 * v + 1] = (_Float16)b1;
    }
    Breg[t] = bm;
    bias[t] = bdtP[col];
  }

  for (int tt = 0; tt < TPW; ++tt) {
    int tile = tile0 + tt;
    int lt = tile % LTILES;
    const float* dtrow =
        dtb + ((size_t)pb * LL + (size_t)lt * 16 + m) * RR;

    // A tile: 16 x 32, only K<6 valid (K pairs are even -> pairs are either
    // fully valid or fully zero)
    v16h a;
#pragma unroll
    for (int v = 0; v < 8; ++v) {
      int k0 = hi * 16 + 2 * v;
      float2 f = {0.f, 0.f};
      if (k0 < RR) f = *(const float2*)(dtrow + k0);
      a[2 * v]     = (_Float16)f.x;
      a[2 * v + 1] = (_Float16)f.y;
    }

    v8f acc[6];
#pragma unroll
    for (int t = 0; t < 6; ++t) acc[t] = (v8f){};
#pragma unroll
    for (int t = 0; t < 6; ++t)
      acc[t] = __builtin_amdgcn_wmma_f32_16x16x32_f16(false, a, false, Breg[t],
                                                      (short)0, acc[t],
                                                      false, false);

    // bias + softplus, store delta[(pb,l),d]
#pragma unroll
    for (int t = 0; t < 6; ++t) {
      int dcol = t * 16 + n;
#pragma unroll
      for (int v = 0; v < 8; ++v) {
        size_t row = (size_t)pb * LL + (size_t)lt * 16 + v + hi * 8;
        float z = acc[t][v] + bias[t];
        float sp = fmaxf(z, 0.f) + log1pf(__expf(-fabsf(z)));  // softplus
        delta[row * CD + dcol] = sp;
      }
    }
  }
}

// ---------------------------------------------------------------------------
// Kernel 4: sequential selective scan. One block per (path,batch); lane = d.
// 16 states per lane in VGPRs; B(t)/C(t) staged into LDS with async b128
// copies (ASYNCcnt), next chunk hinted with global_prefetch_b8.
// ---------------------------------------------------------------------------
#define CHUNK 64
__global__ __launch_bounds__(CD) void ss2d_scan(const float* __restrict__ U,
                                                const float* __restrict__ delta,
                                                const float* __restrict__ Bcb,
                                                const float* __restrict__ Ccb,
                                                const float* __restrict__ A_log,
                                                const float* __restrict__ Dp,
                                                float* __restrict__ Y) {
  int pb = blockIdx.x;          // 0..15
  int p  = pb / BN;
  int d  = threadIdx.x;         // 0..95

  __shared__ float sB[CHUNK * NS];
  __shared__ float sC[CHUNK * NS];

  float Areg[NS], h[NS];
#pragma unroll
  for (int nn = 0; nn < NS; ++nn) {
    Areg[nn] = -__expf(A_log[((size_t)p * CD + d) * NS + nn]);
    h[nn] = 0.f;
  }
  float dscale = Dp[p * CD + d];
  size_t base = (size_t)pb * LL;

  for (int c0 = 0; c0 < LL; c0 += CHUNK) {
    __syncthreads();   // previous chunk fully consumed before LDS overwrite
    for (int i = threadIdx.x * 4; i < CHUNK * NS; i += CD * 4) {
      async_copy_b128(&Bcb[(base + c0) * NS + i], &sB[i]);
      async_copy_b128(&Ccb[(base + c0) * NS + i], &sC[i]);
    }
    wait_asynccnt0();
    __syncthreads();

    if (c0 + CHUNK < LL) {   // hint next chunk into cache (global_prefetch_b8)
      __builtin_prefetch(&Bcb[(base + c0 + CHUNK) * NS], 0, 0);
      __builtin_prefetch(&Ccb[(base + c0 + CHUNK) * NS], 0, 0);
      __builtin_prefetch(&delta[(base + c0 + CHUNK) * CD], 0, 0);
      __builtin_prefetch(&U[(base + c0 + CHUNK) * CD], 0, 0);
    }

    for (int s = 0; s < CHUNK; ++s) {
      size_t row = base + c0 + s;
      float de = delta[row * CD + d];
      float u  = U[row * CD + d];
      float du = de * u;
      float y  = 0.f;
#pragma unroll
      for (int nn = 0; nn < NS; ++nn) {
        float dA = __expf(de * Areg[nn]);                 // exp(delta*A)
        h[nn] = fmaf(dA, h[nn], du * sB[s * NS + nn]);    // h = dA*h + dBu
        y     = fmaf(h[nn], sC[s * NS + nn], y);          // y += h*C
      }
      Y[row * CD + d] = fmaf(u, dscale, y);               // + u*Dp
    }
  }
}

// ---------------------------------------------------------------------------
// Kernel 5: out[b,c,l] = mean over paths of Y[(p,b,l),c]
// (reference averages WITHOUT restoring orientation -> same scan index l)
// ---------------------------------------------------------------------------
__global__ __launch_bounds__(256) void ss2d_reduce(const float* __restrict__ Y,
                                                   float* __restrict__ out,
                                                   int total) {
  int idx = blockIdx.x * blockDim.x + threadIdx.x;
  if (idx >= total) return;
  int l = idx % LL;
  int t = idx / LL;
  int c = t % CD;
  int b = t / CD;
  float s = 0.f;
#pragma unroll
  for (int p = 0; p < NPATH; ++p)
    s += Y[((size_t)(p * BN + b) * LL + l) * CD + c];
  out[idx] = 0.25f * s;
}

// ---------------------------------------------------------------------------
extern "C" void kernel_launch(void* const* d_in, const int* in_sizes, int n_in,
                              void* d_out, int out_size, void* d_ws, size_t ws_size,
                              hipStream_t stream) {
  const float* x     = (const float*)d_in[0];   // (B,C,H,W)
  const float* A_log = (const float*)d_in[1];   // (4,C,N)
  const float* Dp    = (const float*)d_in[2];   // (4,C)
  const float* Wx    = (const float*)d_in[3];   // (4,C,R+2N)
  const float* Wdt   = (const float*)d_in[4];   // (4,R,C)
  const float* bdt   = (const float*)d_in[5];   // (4,C)
  float* out = (float*)d_out;

  // workspace carve-up (floats). Total ~21.4M floats (~86 MB), fits L2.
  float* ws = (float*)d_ws;
  size_t off = 0;
  float* U      = ws + off; off += (size_t)NPATH * BN * LL * CD;  // 6.29M
  float* dtb    = ws + off; off += (size_t)NPATH * BN * LL * RR;  // 0.39M
  float* Bcb    = ws + off; off += (size_t)NPATH * BN * LL * NS;  // 1.05M
  float* Ccb    = ws + off; off += (size_t)NPATH * BN * LL * NS;  // 1.05M
  float* deltab = ws + off; off += (size_t)NPATH * BN * LL * CD;  // 6.29M
  float* Yb     = ws + off; off += (size_t)NPATH * BN * LL * CD;  // 6.29M

  int totU = NPATH * BN * LL * CD;
  ss2d_permute<<<(totU + 255) / 256, 256, 0, stream>>>(x, U, totU);

  ss2d_xproj<<<NWAVES / 8, 256, 0, stream>>>(U, Wx, dtb, Bcb, Ccb);
  ss2d_dtproj<<<NWAVES / 8, 256, 0, stream>>>(dtb, Wdt, bdt, deltab);

  ss2d_scan<<<NPATH * BN, CD, 0, stream>>>(U, deltab, Bcb, Ccb, A_log, Dp, Yb);

  int totO = BN * CD * LL;
  ss2d_reduce<<<(totO + 255) / 256, 256, 0, stream>>>(Yb, out, totO);
}
